// MultiHeadSelfAttentionKVCache_74990128988315
// MI455X (gfx1250) — matrix operand order repaired and checked by reading, not compiled, not run on hardware
//
#include <hip/hip_runtime.h>
#include <hip/hip_bf16.h>

// ---------------------------------------------------------------------------
// MI455X (gfx1250) MHSA forward. wave32, WMMA bf16 (fp32 accum), TDM staging.
// Problem: BS=2, SEQ=2048, DIM=1024, H=16, Dh=64.
// ---------------------------------------------------------------------------

typedef __bf16 bf16_t;
typedef __attribute__((ext_vector_type(16))) __bf16 v16bf;
typedef __attribute__((ext_vector_type(8)))  __bf16 v8bf;
typedef __attribute__((ext_vector_type(8)))  float  v8f;
typedef __attribute__((ext_vector_type(4)))  unsigned int v4u;
typedef __attribute__((ext_vector_type(8)))  int v8i;
typedef __attribute__((ext_vector_type(4)))  int v4i;

#define SEQ   2048
#define DIM   1024
#define HEADS 16
#define HDIM  64
#define BS    2
#define MROWS (BS * SEQ)   // 4096
#define KSTEP 128          // K-slice staged in LDS per TDM transfer

#if defined(__has_builtin)
#if __has_builtin(__builtin_amdgcn_tensor_load_to_lds) && \
    __has_builtin(__builtin_amdgcn_s_wait_tensorcnt)
#define USE_TDM 1
#endif
#endif
#ifndef USE_TDM
#define USE_TDM 0
#endif

__device__ __forceinline__ v16bf cat8(v8bf lo, v8bf hi) {
    v16bf r;
#pragma unroll
    for (int i = 0; i < 8; ++i) { r[i] = lo[i]; r[i + 8] = hi[i]; }
    return r;
}

__device__ __forceinline__ float redmax16(float v) {
    v = fmaxf(v, __shfl_xor(v, 1, 32));
    v = fmaxf(v, __shfl_xor(v, 2, 32));
    v = fmaxf(v, __shfl_xor(v, 4, 32));
    v = fmaxf(v, __shfl_xor(v, 8, 32));
    return v;
}
__device__ __forceinline__ float redsum16(float v) {
    v += __shfl_xor(v, 1, 32);
    v += __shfl_xor(v, 2, 32);
    v += __shfl_xor(v, 4, 32);
    v += __shfl_xor(v, 8, 32);
    return v;
}

// ---------------------------------------------------------------------------
// fp32 -> bf16 conversion (grid-stride)
// ---------------------------------------------------------------------------
__global__ __launch_bounds__(256)
void cvt_f32_bf16(const float* __restrict__ src, bf16_t* __restrict__ dst, int n) {
    for (int i = blockIdx.x * 256 + threadIdx.x; i < n; i += gridDim.x * 256)
        dst[i] = (bf16_t)src[i];
}

// ---------------------------------------------------------------------------
// Shared 64x64-block GEMM mainloop: C(64x64) += A(64xK) * W(nbase..+63, K)^T.
// 4 waves / block; wave w owns rows mrow0 = mbase + w*16, all waves share the
// B panel (64 rows x KSTEP) staged in LDS by the Tensor Data Mover.
// ---------------------------------------------------------------------------
__device__ __forceinline__ void gemm_block64(
    const bf16_t* __restrict__ A, const bf16_t* __restrict__ W,
    bf16_t* lds, int mrow0, int nbase, int lane, int wid, v8f acc[4])
{
    const int half = lane >> 4;
    const int l16  = lane & 15;
    const bf16_t* arow = A + (size_t)(mrow0 + l16) * DIM + half * 16;

    for (int ko = 0; ko < DIM; ko += KSTEP) {
        __syncthreads();                         // previous panel fully consumed
#if USE_TDM
        if (wid == 0) {
            const unsigned long long ga =
                (unsigned long long)(size_t)(W + (size_t)nbase * DIM + ko);
            const unsigned ldsa = (unsigned)(size_t)lds;   // flat[31:0] = LDS offset
            // D# group 0: count=1 (valid), lds_addr, 57-bit global addr, type=2
            v4u g0 = { 1u, ldsa, (unsigned)ga,
                       (unsigned)((ga >> 32) & 0x1FFFFFFu) | 0x80000000u };
            // D# group 1: data_size=2B; tensor_dim0/1 = 1<<20 (no OOB);
            // tile_dim0=KSTEP, tile_dim1=64; tensor_dim0_stride=DIM
            v8i g1 = { (int)(1u << 16),
                       0,
                       16,                                  // dim0[31:16]
                       (int)(16u | ((unsigned)KSTEP << 16)),// dim1[31:16] | tile0
                       64,                                  // tile_dim1
                       DIM,                                 // stride0 (elems)
                       0, 0 };
            v4i gz4  = { 0, 0, 0, 0 };
            v8i gz8  = { 0, 0, 0, 0, 0, 0, 0, 0 };
            // amdgpu-toolchain (clang-23) 6-arg form:
            // (uint32x4, int32x8, int32x4, int32x4, int32x8, i32 cpol)
            __builtin_amdgcn_tensor_load_to_lds(g0, g1, gz4, gz4, gz8, 0);
            __builtin_amdgcn_s_wait_tensorcnt(0);
        }
#else
        {
            const bf16_t* src = W + (size_t)nbase * DIM + ko;
            for (int i = threadIdx.x; i < 64 * KSTEP / 8; i += blockDim.x) {
                const int r = i / (KSTEP / 8), c = i % (KSTEP / 8);
                ((v8bf*)lds)[i] = *(const v8bf*)(src + (size_t)r * DIM + c * 8);
            }
        }
#endif
        __syncthreads();                         // panel visible to all waves
#pragma unroll
        for (int ks = 0; ks < KSTEP; ks += 32) {
            __builtin_prefetch(arow + ko + ks + 256, 0, 0);
            v16bf a = cat8(*(const v8bf*)(arow + ko + ks),
                           *(const v8bf*)(arow + ko + ks + 8));
#pragma unroll
            for (int t = 0; t < 4; ++t) {
                const bf16_t* bp = lds + (t * 16 + l16) * KSTEP + ks + half * 16;
                v16bf b = *(const v16bf*)bp;
                acc[t] = __builtin_amdgcn_wmma_f32_16x16x32_bf16(
                    false, a, false, b, (short)0, acc[t], false, false);
            }
        }
    }
}

// ---------------------------------------------------------------------------
// Q/K/V projection: out = x @ W.T + b. Q,K written (b,h,s,dh); V written
// TRANSPOSED (b,h,dh,s) so the attention P*V B-fragments are contiguous.
// ---------------------------------------------------------------------------
__global__ __launch_bounds__(128)
void qkv_kernel(const bf16_t* __restrict__ xb,
                const bf16_t* __restrict__ wq, const bf16_t* __restrict__ wk,
                const bf16_t* __restrict__ wv,
                const float* __restrict__ bq, const float* __restrict__ bk,
                const float* __restrict__ bv,
                bf16_t* __restrict__ Qb, bf16_t* __restrict__ Kb,
                bf16_t* __restrict__ Vb)
{
    __shared__ bf16_t ldsB[64 * KSTEP];
    const int lane = threadIdx.x & 31;
    const int wid  = threadIdx.x >> 5;
    const int half = lane >> 4;
    const int l16  = lane & 15;
    const int nbase = blockIdx.x * 64;
    const int mrow0 = blockIdx.y * 64 + wid * 16;
    const int mat   = blockIdx.z;

    const bf16_t* W    = (mat == 0) ? wq : (mat == 1) ? wk : wv;
    const float*  bias = (mat == 0) ? bq : (mat == 1) ? bk : bv;
    bf16_t*       Out  = (mat == 0) ? Qb : (mat == 1) ? Kb : Vb;

    v8f acc[4] = {};
    gemm_block64(xb, W, ldsB, mrow0, nbase, lane, wid, acc);

#pragma unroll
    for (int t = 0; t < 4; ++t) {
        const int e  = nbase + t * 16 + l16;
        const int h  = e >> 6;
        const int dh = e & 63;
        const float bval = bias[e];
#pragma unroll
        for (int r = 0; r < 8; ++r) {
            const int mg   = mrow0 + half * 8 + r;
            const int bidx = mg >> 11;
            const int s    = mg & (SEQ - 1);
            const size_t idx = (mat == 2)
                ? ((((size_t)bidx * HEADS + h) * HDIM) + dh) * SEQ + s   // V^T
                : ((((size_t)bidx * HEADS + h) * SEQ) + s) * HDIM + dh;  // Q,K
            Out[idx] = (bf16_t)(acc[t][r] + bval);
        }
    }
}

// ---------------------------------------------------------------------------
// Attention: per-wave 16-query tile over all 2048 keys in chunks of 32.
// Mask is -1e-9 (NOT -inf) -> every key contributes; faithful to reference.
// S = Q K^T (WMMA), online softmax (shfl_xor row reductions), P through LDS
// (C-layout -> A-layout), O += P V^T-layout (WMMA). Output stored (b,h,dh,s)
// reproducing the reference's swapaxes(-1,-2).reshape scramble.
// ---------------------------------------------------------------------------
__global__ __launch_bounds__(32)
void attn_kernel(const bf16_t* __restrict__ Qb, const bf16_t* __restrict__ Kb,
                 const bf16_t* __restrict__ Vt, bf16_t* __restrict__ Ob)
{
    __shared__ bf16_t lds_p[16 * 32];

    const int lane = threadIdx.x & 31;
    const int half = lane >> 4;
    const int l16  = lane & 15;
    const int qbase = blockIdx.x * 16;
    const int h     = blockIdx.y;
    const int bidx  = blockIdx.z;

    const size_t headoff = ((size_t)bidx * HEADS + h) * SEQ * HDIM;
    const bf16_t* Qh = Qb + headoff;
    const bf16_t* Kh = Kb + headoff;
    const bf16_t* Vh = Vt + headoff;   // (dh, s) layout within head

    v16bf qa[2];
#pragma unroll
    for (int c = 0; c < 2; ++c) {
        const bf16_t* p0 = Qh + (size_t)(qbase + l16) * HDIM + c * 32 + half * 16;
        qa[c] = cat8(*(const v8bf*)p0, *(const v8bf*)(p0 + 8));
    }

    const float scale = 0.125f;   // 64^-0.5
    float rmax[8], rsum[8];
#pragma unroll
    for (int r = 0; r < 8; ++r) { rmax[r] = -1.0e30f; rsum[r] = 0.0f; }
    v8f o[4] = {};

    for (int kb = 0; kb < SEQ; kb += 32) {
        float sv[2][8];
#pragma unroll
        for (int t = 0; t < 2; ++t) {
            const int kbase = kb + t * 16;
            v8f sacc = {};
#pragma unroll
            for (int c = 0; c < 2; ++c) {
                const bf16_t* bp =
                    Kh + (size_t)(kbase + l16) * HDIM + c * 32 + half * 16;
                v16bf kf = *(const v16bf*)bp;
                sacc = __builtin_amdgcn_wmma_f32_16x16x32_bf16(
                    false, qa[c], false, kf, (short)0, sacc, false, false);
            }
            const int ng = kbase + l16;
#pragma unroll
            for (int r = 0; r < 8; ++r) {
                const int mg = qbase + half * 8 + r;
                const float v = sacc[r] * scale;
                sv[t][r] = (ng > mg) ? -1e-9f : v;   // faithful tiny-mask bug
            }
        }
        float corr[8];
#pragma unroll
        for (int r = 0; r < 8; ++r) {
            float m  = redmax16(fmaxf(sv[0][r], sv[1][r]));
            float nm = fmaxf(rmax[r], m);
            corr[r] = __expf(rmax[r] - nm);
            rmax[r] = nm;
        }
#pragma unroll
        for (int t = 0; t < 2; ++t)
#pragma unroll
            for (int r = 0; r < 8; ++r)
                sv[t][r] = __expf(sv[t][r] - rmax[r]);
#pragma unroll
        for (int r = 0; r < 8; ++r)
            rsum[r] = rsum[r] * corr[r] + redsum16(sv[0][r] + sv[1][r]);
#pragma unroll
        for (int tt = 0; tt < 4; ++tt)
#pragma unroll
            for (int r = 0; r < 8; ++r) o[tt][r] *= corr[r];

        // P: C-layout -> LDS row-major -> A-layout
#pragma unroll
        for (int t = 0; t < 2; ++t)
#pragma unroll
            for (int r = 0; r < 8; ++r)
                lds_p[(half * 8 + r) * 32 + t * 16 + l16] = (bf16_t)sv[t][r];
        __syncthreads();
        const bf16_t* pp = lds_p + l16 * 32 + half * 16;
        v16bf pf = cat8(*(const v8bf*)pp, *(const v8bf*)(pp + 8));
        __syncthreads();

        // O += P(16x32) x V(32x64); V^T layout -> contiguous b128 loads
#pragma unroll
        for (int tt = 0; tt < 4; ++tt) {
            const bf16_t* vp =
                Vh + (size_t)(tt * 16 + l16) * SEQ + kb + half * 16;
            v16bf vf = *(const v16bf*)vp;
            o[tt] = __builtin_amdgcn_wmma_f32_16x16x32_bf16(
                false, pf, false, vf, (short)0, o[tt], false, false);
        }
    }

    bf16_t* Oh = Ob + ((size_t)bidx * HEADS + h) * HDIM * SEQ;
#pragma unroll
    for (int tt = 0; tt < 4; ++tt) {
        const int dh = tt * 16 + l16;
#pragma unroll
        for (int r = 0; r < 8; ++r) {
            const int mg = qbase + half * 8 + r;
            Oh[(size_t)dh * SEQ + mg] = (bf16_t)(o[tt][r] / rsum[r]);
        }
    }
}

// ---------------------------------------------------------------------------
// Output projection: Y = A @ Wo.T + bo, fp32 out. A = scrambled attention
// buffer viewed as (BS*SEQ, DIM) row-major.
// ---------------------------------------------------------------------------
__global__ __launch_bounds__(128)
void out_kernel(const bf16_t* __restrict__ Ab, const bf16_t* __restrict__ wo,
                const float* __restrict__ bo, float* __restrict__ Y)
{
    __shared__ bf16_t ldsB[64 * KSTEP];
    const int lane = threadIdx.x & 31;
    const int wid  = threadIdx.x >> 5;
    const int half = lane >> 4;
    const int l16  = lane & 15;
    const int nbase = blockIdx.x * 64;
    const int mrow0 = blockIdx.y * 64 + wid * 16;

    v8f acc[4] = {};
    gemm_block64(Ab, wo, ldsB, mrow0, nbase, lane, wid, acc);

#pragma unroll
    for (int t = 0; t < 4; ++t) {
        const int e = nbase + t * 16 + l16;
        const float bval = bo[e];
#pragma unroll
        for (int r = 0; r < 8; ++r) {
            const int mg = mrow0 + half * 8 + r;
            Y[(size_t)mg * DIM + e] = acc[t][r] + bval;
        }
    }
}

// ---------------------------------------------------------------------------
// Launch
// ---------------------------------------------------------------------------
extern "C" void kernel_launch(void* const* d_in, const int* in_sizes, int n_in,
                              void* d_out, int out_size, void* d_ws, size_t ws_size,
                              hipStream_t stream) {
    (void)in_sizes; (void)n_in; (void)out_size; (void)ws_size;

    const float* x  = (const float*)d_in[0];
    // d_in[1] = masks: tril mask applied analytically (-1e-9 fill, as in ref)
    const float* Wq = (const float*)d_in[2];
    const float* bq = (const float*)d_in[3];
    const float* Wk = (const float*)d_in[4];
    const float* bk = (const float*)d_in[5];
    const float* Wv = (const float*)d_in[6];
    const float* bv = (const float*)d_in[7];
    const float* Wo = (const float*)d_in[8];
    const float* bo = (const float*)d_in[9];
    float* out = (float*)d_out;

    char* ws = (char*)d_ws;
    bf16_t* xb  = (bf16_t*)ws;                         //  8 MB: x bf16
    bf16_t* wqb = (bf16_t*)(ws + (8u  << 20));         //  2 MB each
    bf16_t* wkb = wqb + (1u << 20);
    bf16_t* wvb = wkb + (1u << 20);
    bf16_t* wob = wvb + (1u << 20);                    // ends at 16 MB
    bf16_t* Qb  = (bf16_t*)(ws + (16u << 20));         //  8 MB each
    bf16_t* Kb  = Qb + (4u << 20);
    bf16_t* Vb  = Kb + (4u << 20);                     // V stored transposed
    bf16_t* Ab  = Vb + (4u << 20);                     // 40..48 MB

    const int NX = MROWS * DIM;      // 4,194,304
    const int NW = DIM * DIM;        // 1,048,576

    cvt_f32_bf16<<<NX / 256, 256, 0, stream>>>(x,  xb,  NX);
    cvt_f32_bf16<<<NW / 256, 256, 0, stream>>>(Wq, wqb, NW);
    cvt_f32_bf16<<<NW / 256, 256, 0, stream>>>(Wk, wkb, NW);
    cvt_f32_bf16<<<NW / 256, 256, 0, stream>>>(Wv, wvb, NW);
    cvt_f32_bf16<<<NW / 256, 256, 0, stream>>>(Wo, wob, NW);

    qkv_kernel<<<dim3(DIM / 64, MROWS / 64, 3), 128, 0, stream>>>(
        xb, wqb, wkb, wvb, bq, bk, bv, Qb, Kb, Vb);

    attn_kernel<<<dim3(SEQ / 16, HEADS, BS), 32, 0, stream>>>(Qb, Kb, Vb, Ab);

    out_kernel<<<dim3(DIM / 64, MROWS / 64), 128, 0, stream>>>(Ab, wob, bo, out);
}
